// liGRU_70832600646057
// MI455X (gfx1250) — matrix operand order repaired and checked by reading, not compiled, not run on hardware
//
#include <hip/hip_runtime.h>
#include <hip/hip_bf16.h>

typedef __attribute__((ext_vector_type(16))) __bf16 v16bf;
typedef __attribute__((ext_vector_type(8)))  float  v8f;

#define T_STEPS 2000
#define BATCH   32
#define DIN     512
#define UNITS   1024
#define PKEEP   0.8f
#define BNEPS   1e-5f

#define ROWS    (T_STEPS*BATCH)   // 64000
#define MT1     (ROWS/16)         // 4000 M-tiles for input GEMM
#define KT1     (DIN/32)          // 16 k-tiles (K=512)
#define NT      (UNITS/16)        // 64 n-tiles
#define KT2     (UNITS/32)        // 32 k-tiles (K=1024)
#define RB_CNT  (ROWS/256)        // 250 row-blocks for BN partials
#define REC_BLOCKS 8

__device__ __forceinline__ v8f wmma_bf16(v16bf a, v16bf b, v8f c) {
  return __builtin_amdgcn_wmma_f32_16x16x32_bf16(false, a, false, b, (short)0, c, false, false);
}

// ---------------- packing kernels: build exact wave32 WMMA fragment layouts ----------------
// A 16x32 bf16 fragment: lane l(0-15): M=l, elems i<8 -> K=i, i>=8 -> K=i+8; lanes 16-31: +8 on K.
// B 32x16 bf16 fragment: lane l: N=l&15, same K order as A (i<8 -> K=i, i>=8 -> K=i+8; lanes 16-31: +8 on K).

__global__ void pack_x(const float* __restrict__ x, __bf16* __restrict__ xp) {
  int g = blockIdx.x * 256 + threadIdx.x;            // (mtile, ktile, lane)
  int lane  = g & 31;
  int ktile = (g >> 5) & (KT1 - 1);
  int mtile = g >> 9;
  int row = mtile * 16 + (lane & 15);
  int khi = (lane >> 4) << 3;
  v16bf v;
#pragma unroll
  for (int i = 0; i < 16; ++i) {
    int k = ktile * 32 + ((i < 8) ? i : i + 8) + khi;
    v[i] = (__bf16)(x[(size_t)row * DIN + k] * PKEEP);
  }
  *(v16bf*)(xp + (size_t)g * 16) = v;
}

__global__ void pack_w(const float* __restrict__ Wh, const float* __restrict__ Wz,
                       __bf16* __restrict__ wp) {
  int g = blockIdx.x * 256 + threadIdx.x;            // (mat, ntile, ktile, lane)
  int lane  = g & 31;
  int ktile = (g >> 5) & (KT1 - 1);
  int ntile = (g >> 9) & (NT - 1);
  int mat   = g >> 15;
  const float* W = mat ? Wz : Wh;
  int u = ntile * 16 + (lane & 15);
  int kb = ktile * 32 + ((lane >> 4) << 3);
  v16bf v;
#pragma unroll
  for (int i = 0; i < 16; ++i) v[i] = (__bf16)W[(size_t)u * DIN + kb + ((i < 8) ? i : i + 8)];
  *(v16bf*)(wp + (size_t)g * 16) = v;
}

__global__ void pack_u(const float* __restrict__ Uh, const float* __restrict__ Uz,
                       __bf16* __restrict__ up) {
  int g = blockIdx.x * 256 + threadIdx.x;            // (mat, ntile, ktile, lane)
  int lane  = g & 31;
  int ktile = (g >> 5) & (KT2 - 1);
  int ntile = (g >> 10) & (NT - 1);
  int mat   = g >> 16;
  const float* U = mat ? Uz : Uh;
  int u = ntile * 16 + (lane & 15);
  int kb = ktile * 32 + ((lane >> 4) << 3);
  v16bf v;
#pragma unroll
  for (int i = 0; i < 16; ++i) v[i] = (__bf16)U[(size_t)u * UNITS + kb + ((i < 8) ? i : i + 8)];
  *(v16bf*)(up + (size_t)g * 16) = v;
}

// ---------------- phase 1: input GEMMs via WMMA ----------------
__global__ void gemm_in(const __bf16* __restrict__ xp, const __bf16* __restrict__ wp,
                        __bf16* __restrict__ whraw, __bf16* __restrict__ wzraw) {
  int wid  = blockIdx.x * 8 + (threadIdx.x >> 5);
  int lane = threadIdx.x & 31;
  int ntile = wid & (NT - 1);
  int mtile = (wid >> 6) % MT1;
  int mat   = wid / (NT * MT1);
  v8f acc = {};
  const __bf16* ap = xp + ((size_t)mtile * KT1) * 32 * 16 + (size_t)lane * 16;
  const __bf16* bp = wp + ((size_t)(mat * NT + ntile) * KT1) * 32 * 16 + (size_t)lane * 16;
  for (int kt = 0; kt < KT1; ++kt) {
    v16bf a = *(const v16bf*)(ap + (size_t)kt * 512);
    v16bf b = *(const v16bf*)(bp + (size_t)kt * 512);
    acc = wmma_bf16(a, b, acc);
  }
  __bf16* out = mat ? wzraw : whraw;
  int u = ntile * 16 + (lane & 15);
  int mofs = (lane >> 4) << 3;
#pragma unroll
  for (int r = 0; r < 8; ++r) {
    int row = mtile * 16 + mofs + r;
    out[(size_t)row * UNITS + u] = (__bf16)acc[r];
  }
}

// ---------------- BN: two-pass column mean/var, coalesced ----------------
__global__ void bn_part(const __bf16* __restrict__ whraw, const __bf16* __restrict__ wzraw,
                        float* __restrict__ psum, float* __restrict__ psq) {
  int mat = blockIdx.x / RB_CNT;
  int rb  = blockIdx.x % RB_CNT;
  const __bf16* src = mat ? wzraw : whraw;
  int tid = threadIdx.x;
  float s[4] = {0, 0, 0, 0}, q[4] = {0, 0, 0, 0};
  int row0 = rb * 256;
  for (int rr = 0; rr < 256; ++rr) {
    const __bf16* rp = src + (size_t)(row0 + rr) * UNITS;
#pragma unroll
    for (int j = 0; j < 4; ++j) {
      float v = (float)rp[tid + j * 256];
      s[j] += v; q[j] += v * v;
    }
  }
  size_t base = (size_t)(mat * RB_CNT + rb) * UNITS;
#pragma unroll
  for (int j = 0; j < 4; ++j) {
    psum[base + tid + j * 256] = s[j];
    psq [base + tid + j * 256] = q[j];
  }
}

__global__ void bn_final(const float* __restrict__ psum, const float* __restrict__ psq,
                         const float* __restrict__ gwh, const float* __restrict__ bwh,
                         const float* __restrict__ gwz, const float* __restrict__ bwz,
                         float* __restrict__ bnp) {
  int mat = blockIdx.x >> 10;
  int u   = blockIdx.x & (UNITS - 1);
  __shared__ float ss[256], sq[256];
  int tid = threadIdx.x;
  float s = 0.f, q = 0.f;
  for (int rb = tid; rb < RB_CNT; rb += 256) {
    size_t idx = (size_t)(mat * RB_CNT + rb) * UNITS + u;
    s += psum[idx]; q += psq[idx];
  }
  ss[tid] = s; sq[tid] = q; __syncthreads();
  for (int off = 128; off > 0; off >>= 1) {
    if (tid < off) { ss[tid] += ss[tid + off]; sq[tid] += sq[tid + off]; }
    __syncthreads();
  }
  if (tid == 0) {
    float mean = ss[0] / (float)ROWS;
    float var  = sq[0] / (float)ROWS - mean * mean;
    float g  = mat ? gwz[u] : gwh[u];
    float be = mat ? bwz[u] : bwh[u];
    float scale = g * rsqrtf(var + BNEPS);
    bnp[mat * 2048 + u]        = scale;
    bnp[mat * 2048 + 1024 + u] = be - mean * scale;
  }
}

__global__ void init_state(float* __restrict__ hbufs, unsigned* __restrict__ bar) {
  int i = blockIdx.x * 256 + threadIdx.x;
  if (i < 2 * BATCH * UNITS) hbufs[i] = 0.f;
  if (i < 2) bar[i] = 0u;
}

// ---------------- recurrence: persistent cooperative WMMA kernel ----------------
__device__ __forceinline__ void grid_sync(unsigned* bar, unsigned nb) {
  __threadfence();
  __syncthreads();
  if (threadIdx.x == 0) {
    unsigned g = __hip_atomic_load(bar + 1, __ATOMIC_RELAXED, __HIP_MEMORY_SCOPE_AGENT);
    unsigned prev = __hip_atomic_fetch_add(bar, 1u, __ATOMIC_ACQ_REL, __HIP_MEMORY_SCOPE_AGENT);
    if (prev == nb - 1u) {
      __hip_atomic_store(bar, 0u, __ATOMIC_RELAXED, __HIP_MEMORY_SCOPE_AGENT);
      __hip_atomic_fetch_add(bar + 1, 1u, __ATOMIC_ACQ_REL, __HIP_MEMORY_SCOPE_AGENT);
    } else {
      while (__hip_atomic_load(bar + 1, __ATOMIC_ACQUIRE, __HIP_MEMORY_SCOPE_AGENT) == g)
        __builtin_amdgcn_s_sleep(1);
    }
  }
  __syncthreads();
  __threadfence();
}

__global__ void __launch_bounds__(256, 1) ligru_rec(
    const __bf16* __restrict__ up,
    const __bf16* __restrict__ whraw, const __bf16* __restrict__ wzraw,
    const float* __restrict__ bnp,
    float* __restrict__ h0, float* __restrict__ h1,
    float* __restrict__ out, unsigned* __restrict__ bar) {
  __shared__ __attribute__((aligned(32))) __bf16 hA[2][KT2][32][16];  // 64 KB A-fragments
  int tid  = threadIdx.x;
  int lane = tid & 31;
  int wid  = blockIdx.x * 8 + (tid >> 5);          // 0..63 == owned n-tile
  int u = wid * 16 + (lane & 15);
  float scaleH = bnp[u],        shiftH = bnp[1024 + u];
  float scaleZ = bnp[2048 + u], shiftZ = bnp[3072 + u];
  const __bf16* bh = up + ((size_t)(0 * NT + wid) * KT2) * 32 * 16 + (size_t)lane * 16;
  const __bf16* bz = up + ((size_t)(1 * NT + wid) * KT2) * 32 * 16 + (size_t)lane * 16;
  int mofs = (lane >> 4) << 3;

  for (int t = 0; t < T_STEPS; ++t) {
    float* hcur = (t & 1) ? h1 : h0;
    float* hnxt = (t & 1) ? h0 : h1;

    // stage h*PKEEP into LDS as bf16 A-fragments (whole block cooperatively)
    for (int g = tid; g < 2 * KT2 * 32; g += 256) {
      int lane_s = g & 31;
      int ktile  = (g >> 5) & (KT2 - 1);
      int mtile  = g >> 10;
      int b = mtile * 16 + (lane_s & 15);
      int kh = (lane_s >> 4) << 3;
      const float* hp = hcur + (size_t)b * UNITS + ktile * 32;
      v16bf v;
#pragma unroll
      for (int i = 0; i < 16; ++i) {
        int k = ((i < 8) ? i : i + 8) + kh;
        v[i] = (__bf16)(hp[k] * PKEEP);
      }
      *(v16bf*)&hA[mtile][ktile][lane_s][0] = v;
    }
    __syncthreads();

    v8f aH0 = {}, aH1 = {}, aZ0 = {}, aZ1 = {};
    for (int kt = 0; kt < KT2; ++kt) {
      v16bf a0  = *(const v16bf*)&hA[0][kt][lane][0];
      v16bf a1  = *(const v16bf*)&hA[1][kt][lane][0];
      v16bf vbh = *(const v16bf*)(bh + (size_t)kt * 512);
      v16bf vbz = *(const v16bf*)(bz + (size_t)kt * 512);
      aH0 = wmma_bf16(a0, vbh, aH0);
      aH1 = wmma_bf16(a1, vbh, aH1);
      aZ0 = wmma_bf16(a0, vbz, aZ0);
      aZ1 = wmma_bf16(a1, vbz, aZ1);
    }

    size_t rowbase = (size_t)t * BATCH;
#pragma unroll
    for (int m = 0; m < 2; ++m) {
      const v8f accH = m ? aH1 : aH0;
      const v8f accZ = m ? aZ1 : aZ0;
#pragma unroll
      for (int r = 0; r < 8; ++r) {
        int b = m * 16 + mofs + r;
        size_t idx = (rowbase + b) * UNITS + u;
        float wz = (float)wzraw[idx] * scaleZ + shiftZ + accZ[r];
        float z  = 1.0f / (1.0f + __expf(-wz));
        float wh = (float)whraw[idx] * scaleH + shiftH + accH[r];
        float hc = fmaxf(wh, 0.0f) * PKEEP;
        float hold = hcur[(size_t)b * UNITS + u];
        float hnew = z * hold + (1.0f - z) * hc;
        hnxt[(size_t)b * UNITS + u] = hnew;
        out[idx] = hnew;
      }
    }
    if (t + 1 < T_STEPS) {
      __builtin_prefetch(whraw + (rowbase + BATCH) * UNITS + u, 0, 1);
      __builtin_prefetch(wzraw + (rowbase + BATCH) * UNITS + u, 0, 1);
    }
    grid_sync(bar, REC_BLOCKS);
  }
}

extern "C" void kernel_launch(void* const* d_in, const int* in_sizes, int n_in,
                              void* d_out, int out_size, void* d_ws, size_t ws_size,
                              hipStream_t stream) {
  (void)in_sizes; (void)n_in; (void)out_size; (void)ws_size;
  const float* x   = (const float*)d_in[0];
  const float* Wh  = (const float*)d_in[1];
  const float* Wz  = (const float*)d_in[2];
  const float* Uh  = (const float*)d_in[3];
  const float* Uz  = (const float*)d_in[4];
  const float* gwh = (const float*)d_in[5];
  const float* bwh = (const float*)d_in[6];
  const float* gwz = (const float*)d_in[7];
  const float* bwz = (const float*)d_in[8];
  float* out = (float*)d_out;

  char* p = (char*)d_ws;
  auto alloc = [&](size_t bytes) -> char* {
    char* r = p;
    p += (bytes + 255) & ~(size_t)255;
    return r;
  };
  __bf16* xp    = (__bf16*)alloc((size_t)MT1 * KT1 * 32 * 16 * 2);      // 65.5 MB
  __bf16* wp    = (__bf16*)alloc((size_t)2 * NT * KT1 * 32 * 16 * 2);   // 2 MB
  __bf16* up    = (__bf16*)alloc((size_t)2 * NT * KT2 * 32 * 16 * 2);   // 4 MB
  __bf16* whraw = (__bf16*)alloc((size_t)ROWS * UNITS * 2);             // 131 MB
  __bf16* wzraw = (__bf16*)alloc((size_t)ROWS * UNITS * 2);             // 131 MB
  float* psum   = (float*)alloc((size_t)2 * RB_CNT * UNITS * 4);
  float* psq    = (float*)alloc((size_t)2 * RB_CNT * UNITS * 4);
  float* bnp    = (float*)alloc(4096 * 4);
  float* hb     = (float*)alloc((size_t)2 * BATCH * UNITS * 4);
  unsigned* bar = (unsigned*)alloc(256);

  pack_x<<<MT1 * KT1 * 32 / 256, 256, 0, stream>>>(x, xp);
  pack_w<<<2 * NT * KT1 * 32 / 256, 256, 0, stream>>>(Wh, Wz, wp);
  pack_u<<<2 * NT * KT2 * 32 / 256, 256, 0, stream>>>(Uh, Uz, up);
  gemm_in<<<2 * MT1 * NT / 8, 256, 0, stream>>>(xp, wp, whraw, wzraw);
  bn_part<<<2 * RB_CNT, 256, 0, stream>>>(whraw, wzraw, psum, psq);
  bn_final<<<2 * UNITS, 256, 0, stream>>>(psum, psq, gwh, bwh, gwz, bwz, bnp);
  init_state<<<(2 * BATCH * UNITS + 255) / 256, 256, 0, stream>>>(hb, bar);
  ligru_rec<<<REC_BLOCKS, 256, 0, stream>>>(up, whraw, wzraw, bnp,
                                            hb, hb + BATCH * UNITS, out, bar);
}
